// SRT_53214644798202
// MI455X (gfx1250) — matrix operand order repaired and checked by reading, not compiled, not run on hardware
//
#include <hip/hip_runtime.h>
#include <hip/hip_bf16.h>
#include <math.h>

// ---------------------------------------------------------------------------
// CDNA5 (gfx1250) Star-Transformer forward.
// Dense GEMMs: v_wmma_f32_16x16x32_bf16 with double-buffered
// global_load_async_to_lds_b128 staging of the activation (B) operand.
// ---------------------------------------------------------------------------

typedef __bf16 bf16_t;
typedef __attribute__((ext_vector_type(16))) __bf16 v16bf;
typedef __attribute__((ext_vector_type(8)))  __bf16 v8bf;
typedef __attribute__((ext_vector_type(8)))  float  v8f;

#define C_DIM     256
#define DIN       1024
#define V_NODES   17
#define T_DIM     256
#define B_DIM     4
#define N_BT      1024                    // B*T graph instances
#define R_ROWS    (N_BT * V_NODES)        // 17408 node rows
#define R18_ROWS  (N_BT * 18)             // 18432 sru key rows
#define INV_SQRT_DK 0.17677669529663687f  // 1/sqrt(32)

#define F_RELU 1
#define F_RES  2

#define BSTRIDE 80   // LDS bytes per staged column: 64B data + 16B pad (bank spread)

// graph adjacency (0-indexed), padded
__constant__ int d_nbr[17][5] = {
  {0,1,2,5,6},{0,1,3,0,0},{0,2,4,0,0},{1,3,0,0,0},{2,4,0,0,0},
  {0,5,7,11,0},{0,6,8,12,0},{5,7,9,0,0},{6,8,12,0,0},{7,9,0,0,0},
  {8,10,0,0,0},{8,11,13,0,0},{10,12,14,0,0},{11,13,15,0,0},{12,14,16,0,0},
  {13,15,0,0,0},{14,16,0,0,0}};
__constant__ int d_nnb[17] = {5,3,3,2,2,4,4,3,3,2,2,3,3,3,3,2,2};

__device__ __forceinline__ float wave_sum(float x) {
#pragma unroll
  for (int o = 16; o > 0; o >>= 1) x += __shfl_xor(x, o, 32);
  return x;
}

// ---------------------------------------------------------------------------
// Weight swizzle: row-major fp32 W[M][K] -> per-A-fragment bf16 layout.
// Tile (mt,kt) is 16x32; lane l<16 holds row m=mt*16+l, K = {0..7,16..23}+kt*32;
// lane l>=16 holds same row set, K = {8..15,24..31}+kt*32 (ISA 7.12.2).
// ---------------------------------------------------------------------------
__global__ __launch_bounds__(32) void swizzle_w(const float* __restrict__ W,
                                                bf16_t* __restrict__ out,
                                                int K, int KT) {
  const int tile = blockIdx.x;
  const int lane = threadIdx.x;
  const int mt = tile / KT, kt = tile % KT;
  const int m = mt * 16 + (lane & 15);
  const int kbase = kt * 32 + (lane >> 4) * 8;
  bf16_t* o = out + (size_t)tile * 512 + lane * 16;
  const float* w = W + (size_t)m * K + kbase;
#pragma unroll
  for (int j = 0; j < 16; ++j) {
    int k = (j & 7) + ((j >> 3) * 16);
    o[j] = (bf16_t)w[k];
  }
}

// ---------------------------------------------------------------------------
// GEMM: Y[n][m] = act( sum_k W[m][k]*X[n][k] + bias[m] (+ Res[n][m]) )
// A: swizzled bf16 weights (global, b128 loads + WGP-scope prefetch).
// B: (ncols,K) bf16, staged per 64x32 K-tile into LDS by the async engine
//    (double-buffered), fragments read back with ds_load_b128.
// Block 256 thr = 8 waves; block tile 128(M) x 64(N); wave tile 32x32.
// ---------------------------------------------------------------------------
__device__ __forceinline__ void store_frag(v8f c, int mb, int nb, int ll, int lh,
                                           int M, const float* __restrict__ bias,
                                           float* __restrict__ Yf,
                                           bf16_t* __restrict__ Ybf,
                                           const float* __restrict__ Res,
                                           int flags) {
  const int n = nb + ll;
  const int m = mb + lh * 8;            // C-frag: lanes 0-15 rows m..m+7, 16-31 rows m+8..m+15
  const size_t base = (size_t)n * M + m;
#pragma unroll
  for (int r = 0; r < 8; ++r) {
    float val = c[r] + bias[m + r];
    if (flags & F_RES) val += Res[base + r];
    if (flags & F_RELU) val = fmaxf(val, 0.f);
    if (Yf)  Yf[base + r] = val;
    if (Ybf) Ybf[base + r] = (bf16_t)val;
  }
}

__global__ __launch_bounds__(256) void gemm_bf16(const bf16_t* __restrict__ A,
                                                 const float* __restrict__ bias,
                                                 const bf16_t* __restrict__ X,
                                                 float* __restrict__ Yf,
                                                 bf16_t* __restrict__ Ybf,
                                                 const float* __restrict__ Res,
                                                 int M, int K, int flags) {
  __shared__ __align__(32) char sB[2][64 * BSTRIDE];

  const int tid = threadIdx.x;
  const int lane = tid & 31;
  const int wave = tid >> 5;
  const int wm = wave & 3, wn = wave >> 2;
  const int m0 = blockIdx.x * 128 + wm * 32;
  const int n0 = blockIdx.y * 64 + wn * 32;
  const int KT = K >> 5;
  const int ll = lane & 15, lh = lane >> 4;

  // cooperative async stage: thread copies 16B of the 64x64B tile
  const int ccol = tid >> 2, cpc = tid & 3;                 // col 0..63, piece 0..3
  const char* gsrc =
      (const char*)(X + (size_t)(blockIdx.y * 64 + ccol) * K) + cpc * 16;
  unsigned ldsdst0 = (unsigned)(uintptr_t)(&sB[0][ccol * BSTRIDE + cpc * 16]);
  unsigned ldsdst1 = (unsigned)(uintptr_t)(&sB[1][ccol * BSTRIDE + cpc * 16]);

  auto stage = [&](int kt, int p) {
    unsigned dst = p ? ldsdst1 : ldsdst0;
    asm volatile("global_load_async_to_lds_b128 %0, %1, off"
                 :: "v"(dst), "v"(gsrc + (size_t)kt * 64) : "memory");
  };

  stage(0, 0);
  asm volatile("s_wait_asynccnt 0x0" ::: "memory");
  __syncthreads();

  v8f c00 = {}, c01 = {}, c10 = {}, c11 = {};
  const bf16_t* aB0 = A + (size_t)(m0 >> 4) * KT * 512 + lane * 16;
  const bf16_t* aB1 = aB0 + (size_t)KT * 512;
  // B-frag lane ll holds col (n0+ll), K = {0..7,16..23} (lanes<16) / {8..15,24..31} (lanes>=16)
  const char* bCol0 = &sB[0][(wn * 32 + ll) * BSTRIDE + lh * 16];
  const char* bCol1 = &sB[1][(wn * 32 + ll) * BSTRIDE + lh * 16];

  for (int kt = 0; kt < KT; ++kt) {
    const int p = kt & 1;
    if (kt + 1 < KT) stage(kt + 1, p ^ 1);
    if (kt + 4 < KT) {          // WGP-scope prefetch of the A stream (loc=3)
      __builtin_prefetch(aB0 + (size_t)(kt + 4) * 512, 0, 3);
      __builtin_prefetch(aB1 + (size_t)(kt + 4) * 512, 0, 3);
    }
    v16bf a0 = *(const v16bf*)(aB0 + (size_t)kt * 512);
    v16bf a1 = *(const v16bf*)(aB1 + (size_t)kt * 512);
    const char* bp = p ? bCol1 : bCol0;
    v8bf b0lo = *(const v8bf*)(bp);
    v8bf b0hi = *(const v8bf*)(bp + 32);
    v8bf b1lo = *(const v8bf*)(bp + 16 * BSTRIDE);
    v8bf b1hi = *(const v8bf*)(bp + 16 * BSTRIDE + 32);
    v16bf b0 = __builtin_shufflevector(b0lo, b0hi, 0, 1, 2, 3, 4, 5, 6, 7,
                                       8, 9, 10, 11, 12, 13, 14, 15);
    v16bf b1 = __builtin_shufflevector(b1lo, b1hi, 0, 1, 2, 3, 4, 5, 6, 7,
                                       8, 9, 10, 11, 12, 13, 14, 15);
    c00 = __builtin_amdgcn_wmma_f32_16x16x32_bf16(false, a0, false, b0, (short)0, c00, false, false);
    c01 = __builtin_amdgcn_wmma_f32_16x16x32_bf16(false, a0, false, b1, (short)0, c01, false, false);
    c10 = __builtin_amdgcn_wmma_f32_16x16x32_bf16(false, a1, false, b0, (short)0, c10, false, false);
    c11 = __builtin_amdgcn_wmma_f32_16x16x32_bf16(false, a1, false, b1, (short)0, c11, false, false);
    if (kt + 1 < KT) {
      asm volatile("s_wait_asynccnt 0x0" ::: "memory");
      __syncthreads();
    }
  }
  store_frag(c00, m0,      n0,      ll, lh, M, bias, Yf, Ybf, Res, flags);
  store_frag(c01, m0,      n0 + 16, ll, lh, M, bias, Yf, Ybf, Res, flags);
  store_frag(c10, m0 + 16, n0,      ll, lh, M, bias, Yf, Ybf, Res, flags);
  store_frag(c11, m0 + 16, n0 + 16, ll, lh, M, bias, Yf, Ybf, Res, flags);
}

// ---------------------------------------------------------------------------
// data (B,C,V,T) -> node rows (bt*17+v, C); also bf16 copy (constant "embs")
// ---------------------------------------------------------------------------
__global__ __launch_bounds__(256) void build_embs(const float* __restrict__ data,
                                                  float* __restrict__ X,
                                                  bf16_t* __restrict__ Ebf) {
  const size_t i = (size_t)blockIdx.x * 256 + threadIdx.x;   // over R_ROWS*C
  const int c = (int)(i & 255);
  size_t r = i >> 8;
  const int v = (int)(r % V_NODES);
  const int bt = (int)(r / V_NODES);
  const int b = bt >> 8, t = bt & 255;
  const float val = data[(((size_t)(b * C_DIM + c)) * V_NODES + v) * T_DIM + t];
  X[i] = val;
  Ebf[i] = (bf16_t)val;
}

__global__ __launch_bounds__(256) void relay_init(const float* __restrict__ X,
                                                  float* __restrict__ relay,
                                                  bf16_t* __restrict__ relay_bf,
                                                  bf16_t* __restrict__ relay0_bf) {
  const int bt = blockIdx.x, c = threadIdx.x;
  float s = 0.f;
#pragma unroll
  for (int v = 0; v < V_NODES; ++v) s += X[((size_t)bt * V_NODES + v) * C_DIM + c];
  s *= (1.f / V_NODES);
  relay[(size_t)bt * C_DIM + c] = s;
  relay_bf[(size_t)bt * C_DIM + c] = (bf16_t)s;
  relay0_bf[(size_t)bt * C_DIM + c] = (bf16_t)s;
}

// LayerNorm over C per row
__global__ __launch_bounds__(256) void ln_rows(const float* __restrict__ X,
                                               const float* __restrict__ g,
                                               const float* __restrict__ b,
                                               float* __restrict__ XN,
                                               bf16_t* __restrict__ XNbf) {
  const int r = blockIdx.x, c = threadIdx.x;
  const float x = X[(size_t)r * C_DIM + c];
  __shared__ float s1[8], s2[8], tot[2];
  float v1 = wave_sum(x), v2 = wave_sum(x * x);
  const int w = threadIdx.x >> 5;
  if ((threadIdx.x & 31) == 0) { s1[w] = v1; s2[w] = v2; }
  __syncthreads();
  if (threadIdx.x == 0) {
    float a = 0.f, q = 0.f;
    for (int i = 0; i < 8; ++i) { a += s1[i]; q += s2[i]; }
    tot[0] = a; tot[1] = q;
  }
  __syncthreads();
  const float mean = tot[0] * (1.f / C_DIM);
  const float var  = tot[1] * (1.f / C_DIM) - mean * mean;
  const float y = (x - mean) * rsqrtf(var + 1e-6f) * g[c] + b[c];
  XN[(size_t)r * C_DIM + c] = y;
  XNbf[(size_t)r * C_DIM + c] = (bf16_t)y;
}

// deterministic per-channel stats (BatchNorm over all rows)
__global__ __launch_bounds__(256) void colstats_part(const float* __restrict__ X,
                                                     float* __restrict__ part) {
  const int c = threadIdx.x;
  const float* p = X + (size_t)blockIdx.x * 64 * C_DIM + c;
  float s = 0.f, q = 0.f;
#pragma unroll 4
  for (int i = 0; i < 64; ++i) { float x = p[(size_t)i * C_DIM]; s += x; q += x * x; }
  part[(size_t)blockIdx.x * 512 + c] = s;
  part[(size_t)blockIdx.x * 512 + C_DIM + c] = q;
}

__global__ __launch_bounds__(256) void colstats_final(const float* __restrict__ part,
                                                      float* __restrict__ stats,
                                                      int chunks, float inv_cnt) {
  const int c = threadIdx.x;
  float s = 0.f, q = 0.f;
  for (int i = 0; i < chunks; ++i) {
    s += part[(size_t)i * 512 + c];
    q += part[(size_t)i * 512 + C_DIM + c];
  }
  const float mean = s * inv_cnt;
  const float var  = q * inv_cnt - mean * mean;
  stats[c] = mean;
  stats[C_DIM + c] = rsqrtf(var + 1e-5f);
}

__global__ __launch_bounds__(256) void bn_norm(const float* __restrict__ X,
                                               const float* __restrict__ stats,
                                               const float* __restrict__ g,
                                               const float* __restrict__ b,
                                               float* __restrict__ Yf,
                                               bf16_t* __restrict__ Ybf, int leaky) {
  const int c = threadIdx.x;
  const size_t idx = (size_t)blockIdx.x * C_DIM + c;
  float y = (X[idx] - stats[c]) * stats[C_DIM + c] * g[c] + b[c];
  if (leaky) y = (y > 0.f) ? y : 0.01f * y;
  Yf[idx] = y;
  Ybf[idx] = (bf16_t)y;
}

// sju attention: per node row, 7 candidates (5 graph nbrs + embs-self + relay)
__global__ __launch_bounds__(256) void attn_sju(const float* __restrict__ Q,
                                                const float* __restrict__ Kn,
                                                const float* __restrict__ Vn,
                                                const float* __restrict__ KE,
                                                const float* __restrict__ VE,
                                                const float* __restrict__ KR,
                                                const float* __restrict__ VR,
                                                float* __restrict__ XN) {
  const int r = blockIdx.x;
  const int bt = r / V_NODES;
  const int v = r - bt * V_NODES;
  const int ch = threadIdx.x;                 // h*32 + d; wave == head
  const float q = Q[(size_t)r * C_DIM + ch];
  const int nn = d_nnb[v];
  float sc[7];
#pragma unroll
  for (int w = 0; w < 5; ++w) {
    float kv = 0.f;
    if (w < nn) kv = Kn[((size_t)bt * V_NODES + d_nbr[v][w]) * C_DIM + ch];
    float s = wave_sum(q * kv);
    sc[w] = (w < nn) ? s * INV_SQRT_DK : -1e30f;
  }
  sc[5] = wave_sum(q * KE[(size_t)r * C_DIM + ch]) * INV_SQRT_DK;
  sc[6] = wave_sum(q * KR[(size_t)bt * C_DIM + ch]) * INV_SQRT_DK;
  float mx = sc[0];
#pragma unroll
  for (int w = 1; w < 7; ++w) mx = fmaxf(mx, sc[w]);
  float e[7], den = 0.f;
#pragma unroll
  for (int w = 0; w < 7; ++w) { e[w] = expf(sc[w] - mx); den += e[w]; }
  float att = 0.f;
#pragma unroll
  for (int w = 0; w < 5; ++w)
    if (w < nn) att += e[w] * Vn[((size_t)bt * V_NODES + d_nbr[v][w]) * C_DIM + ch];
  att += e[5] * VE[(size_t)r * C_DIM + ch];
  att += e[6] * VR[(size_t)bt * C_DIM + ch];
  XN[(size_t)r * C_DIM + ch] += att / den;    // in place: xn + att
}

// sru attention: 18 keys per batch (relay + 17 nodes)
__global__ __launch_bounds__(256) void attn_sru(const float* __restrict__ RQ,
                                                const float* __restrict__ RK,
                                                const float* __restrict__ RV,
                                                const float* __restrict__ RELAY,
                                                float* __restrict__ RATT) {
  const int bt = blockIdx.x;
  const int ch = threadIdx.x;
  const float q = RQ[(size_t)bt * C_DIM + ch];
  float sc[18];
#pragma unroll
  for (int j = 0; j < 18; ++j)
    sc[j] = wave_sum(q * RK[((size_t)bt * 18 + j) * C_DIM + ch]) * INV_SQRT_DK;
  float mx = sc[0];
#pragma unroll
  for (int j = 1; j < 18; ++j) mx = fmaxf(mx, sc[j]);
  float den = 0.f, att = 0.f;
  float e[18];
#pragma unroll
  for (int j = 0; j < 18; ++j) { e[j] = expf(sc[j] - mx); den += e[j]; }
#pragma unroll
  for (int j = 0; j < 18; ++j)
    att += e[j] * RV[((size_t)bt * 18 + j) * C_DIM + ch];
  RATT[(size_t)bt * C_DIM + ch] = RELAY[(size_t)bt * C_DIM + ch] + att / den;
}

__global__ __launch_bounds__(256) void concat_y18(const bf16_t* __restrict__ relay_bf,
                                                  const bf16_t* __restrict__ x_bf,
                                                  bf16_t* __restrict__ y18) {
  const int row = blockIdx.x, c = threadIdx.x;
  const int bt = row / 18, j = row - bt * 18;
  bf16_t val = (j == 0) ? relay_bf[(size_t)bt * C_DIM + c]
                        : x_bf[((size_t)bt * V_NODES + (j - 1)) * C_DIM + c];
  y18[(size_t)row * C_DIM + c] = val;
}

// nodes (bt*17+v, C) -> out (B,C,V,T)
__global__ __launch_bounds__(256) void output_write(const float* __restrict__ X,
                                                    float* __restrict__ out) {
  const size_t i = (size_t)blockIdx.x * 256 + threadIdx.x;
  const int t = (int)(i & 255);
  size_t u = i >> 8;
  const int v = (int)(u % V_NODES);
  const size_t w = u / V_NODES;
  const int c = (int)(w & 255);
  const int b = (int)(w >> 8);
  out[i] = X[(((size_t)(b * T_DIM + t)) * V_NODES + v) * C_DIM + c];
}

// ---------------------------------------------------------------------------
// Workspace arena (bytes), all 256B aligned.  Lifetime-checked aliases below.
// ---------------------------------------------------------------------------
static constexpr size_t ALN(size_t x) { return (x + 255) & ~(size_t)255; }
static constexpr size_t RC   = (size_t)R_ROWS * C_DIM;       // 4,456,448 elems
static constexpr size_t NC   = (size_t)N_BT * C_DIM;
static constexpr size_t R18C = (size_t)R18_ROWS * C_DIM;
static constexpr size_t WL   = 3 * 65536 + 2 * 262144 + 3 * 65536 + 2 * 262144; // elems/layer

static constexpr size_t OFF_WBF     = 0;
static constexpr size_t OFF_EMBSBF  = ALN(OFF_WBF + 2 * WL * 2);
static constexpr size_t OFF_R0BF    = ALN(OFF_EMBSBF + RC * 2);
static constexpr size_t OFF_X       = ALN(OFF_R0BF + NC * 2);
static constexpr size_t OFF_XBF     = ALN(OFF_X + RC * 4);
static constexpr size_t OFF_RELAY   = ALN(OFF_XBF + RC * 2);
static constexpr size_t OFF_RELAYBF = ALN(OFF_RELAY + NC * 4);
static constexpr size_t OFF_PART    = ALN(OFF_RELAYBF + NC * 2);
static constexpr size_t OFF_STATS   = ALN(OFF_PART + 272 * 512 * 4);
static constexpr size_t OFF_XN      = ALN(OFF_STATS + 512 * 4);
static constexpr size_t OFF_XNBF    = OFF_XN + RC * 4;
static constexpr size_t OFF_Q       = OFF_XNBF + RC * 2;
static constexpr size_t OFF_K       = OFF_Q + RC * 4;
static constexpr size_t OFF_V       = OFF_K + RC * 4;     // contiguous with K (H alias)
static constexpr size_t OFF_KE      = OFF_V + RC * 4;
static constexpr size_t OFF_VE      = OFF_KE + RC * 4;
static constexpr size_t OFF_KR      = OFF_VE + RC * 4;
static constexpr size_t OFF_VR      = OFF_KR + NC * 4;
static constexpr size_t OFF_RQ      = OFF_VR + NC * 4;
static constexpr size_t OFF_RATT    = OFF_RQ + NC * 4;
static constexpr size_t OFF_RRET    = OFF_RATT + NC * 4;
static constexpr size_t OFF_RRETBF  = OFF_RRET + NC * 4;
static constexpr size_t OFF_RH      = OFF_RRETBF + NC * 2;
static constexpr size_t OFF_RY2     = OFF_RH + (size_t)N_BT * DIN * 2;
// aliases (dead-buffer reuse, lifetimes verified against launch order):
static constexpr size_t OFF_RET     = OFF_Q;     // after attention consumed Q
static constexpr size_t OFF_RETBF   = OFF_XNBF;  // after q/k/v gemms consumed XNbf
static constexpr size_t OFF_H       = OFF_K;     // bf16, exactly spans K+V (dead)
static constexpr size_t OFF_Y2      = OFF_KE;    // KE dead after attention
static constexpr size_t OFF_Y18BF   = OFF_VE;    // VE dead after attention
static constexpr size_t OFF_RK      = OFF_K;     // H dead after f2 gemm
static constexpr size_t OFF_RV      = OFF_XN;    // XN(att)/XNbf dead by sru time

extern "C" void kernel_launch(void* const* d_in, const int* in_sizes, int n_in,
                              void* d_out, int out_size, void* d_ws, size_t ws_size,
                              hipStream_t stream) {
  const float* in[31];
  for (int i = 0; i < 31 && i < n_in; ++i) in[i] = (const float*)d_in[i];
  const float* data = in[0];
  const float* ln_g = in[1];
  const float* ln_b = in[2];
  // j: 3 q_w,4 q_b,5 k_w,6 k_b,7 v_w,8 v_b,9 bn_g,10 bn_b,11 f1_w,12 f1_b,
  //    13 f2_w,14 f2_b,15 fbn_g,16 fbn_b ; r: same +14

  char* ws = (char*)d_ws;
  auto F  = [&](size_t off) { return (float*)(ws + off); };
  auto BF = [&](size_t off) { return (bf16_t*)(ws + off); };
  bf16_t* WBF = BF(OFF_WBF);

  // ---- per-call weight prep: fp32 -> swizzled bf16 A-fragments ----
  struct WJ { int idx; int M; int K; size_t off; };
  const WJ wj[10] = {
    {3, C_DIM, C_DIM, 0},        {5, C_DIM, C_DIM, 65536},   {7, C_DIM, C_DIM, 131072},
    {11, DIN, C_DIM, 196608},    {13, C_DIM, DIN, 458752},
    {17, C_DIM, C_DIM, 720896},  {19, C_DIM, C_DIM, 786432}, {21, C_DIM, C_DIM, 851968},
    {25, DIN, C_DIM, 917504},    {27, C_DIM, DIN, 1179648}};
  for (int l = 0; l < 2; ++l)
    for (int j = 0; j < 10; ++j) {
      const WJ& t = wj[j];
      int KT = t.K / 32, tiles = (t.M / 16) * KT;
      swizzle_w<<<tiles, 32, 0, stream>>>(in[t.idx] + (size_t)l * t.M * t.K,
                                          WBF + (size_t)l * WL + t.off, t.K, KT);
    }

  build_embs<<<R_ROWS, 256, 0, stream>>>(data, F(OFF_X), BF(OFF_EMBSBF));
  relay_init<<<N_BT, 256, 0, stream>>>(F(OFF_X), F(OFF_RELAY), BF(OFF_RELAYBF), BF(OFF_R0BF));

  auto gemm = [&](const bf16_t* A, const float* bias, const bf16_t* Xb,
                  float* Yf, bf16_t* Ybf, const float* Res,
                  int M, int K, int ncols, int flags) {
    dim3 g(M / 128, ncols / 64);
    gemm_bf16<<<g, 256, 0, stream>>>(A, bias, Xb, Yf, Ybf, Res, M, K, flags);
  };
  float* part = F(OFF_PART);
  float* stats = F(OFF_STATS);
  auto run_stats = [&](const float* Xp, int rows) {
    int chunks = rows / 64;
    colstats_part<<<chunks, 256, 0, stream>>>(Xp, part);
    colstats_final<<<1, 256, 0, stream>>>(part, stats, chunks, 1.0f / rows);
  };

  for (int l = 0; l < 2; ++l) {
    const bf16_t* Wjq = WBF + (size_t)l * WL + 0;
    const bf16_t* Wjk = WBF + (size_t)l * WL + 65536;
    const bf16_t* Wjv = WBF + (size_t)l * WL + 131072;
    const bf16_t* Wf1 = WBF + (size_t)l * WL + 196608;
    const bf16_t* Wf2 = WBF + (size_t)l * WL + 458752;
    const bf16_t* Wrq = WBF + (size_t)l * WL + 720896;
    const bf16_t* Wrk = WBF + (size_t)l * WL + 786432;
    const bf16_t* Wrv = WBF + (size_t)l * WL + 851968;
    const bf16_t* Wg1 = WBF + (size_t)l * WL + 917504;
    const bf16_t* Wg2 = WBF + (size_t)l * WL + 1179648;
    const size_t lc = (size_t)l * C_DIM, ld = (size_t)l * DIN;

    // ---------- sju ----------
    ln_rows<<<R_ROWS, 256, 0, stream>>>(F(OFF_X), ln_g + lc, ln_b + lc,
                                        F(OFF_XN), BF(OFF_XNBF));
    gemm(Wjq, in[4] + lc, BF(OFF_XNBF),   F(OFF_Q),  nullptr, nullptr, C_DIM, C_DIM, R_ROWS, 0);
    gemm(Wjk, in[6] + lc, BF(OFF_XNBF),   F(OFF_K),  nullptr, nullptr, C_DIM, C_DIM, R_ROWS, 0);
    gemm(Wjv, in[8] + lc, BF(OFF_XNBF),   F(OFF_V),  nullptr, nullptr, C_DIM, C_DIM, R_ROWS, 0);
    gemm(Wjk, in[6] + lc, BF(OFF_EMBSBF), F(OFF_KE), nullptr, nullptr, C_DIM, C_DIM, R_ROWS, 0);
    gemm(Wjv, in[8] + lc, BF(OFF_EMBSBF), F(OFF_VE), nullptr, nullptr, C_DIM, C_DIM, R_ROWS, 0);
    gemm(Wjk, in[6] + lc, BF(OFF_R0BF),   F(OFF_KR), nullptr, nullptr, C_DIM, C_DIM, N_BT, 0);
    gemm(Wjv, in[8] + lc, BF(OFF_R0BF),   F(OFF_VR), nullptr, nullptr, C_DIM, C_DIM, N_BT, 0);
    attn_sju<<<R_ROWS, 256, 0, stream>>>(F(OFF_Q), F(OFF_K), F(OFF_V), F(OFF_KE),
                                         F(OFF_VE), F(OFF_KR), F(OFF_VR), F(OFF_XN));
    run_stats(F(OFF_XN), R_ROWS);
    bn_norm<<<R_ROWS, 256, 0, stream>>>(F(OFF_XN), stats, in[9] + lc, in[10] + lc,
                                        F(OFF_RET), BF(OFF_RETBF), 0);
    gemm(Wf1, in[12] + ld, BF(OFF_RETBF), nullptr, BF(OFF_H), nullptr, DIN, C_DIM, R_ROWS, F_RELU);
    gemm(Wf2, in[14] + lc, BF(OFF_H), F(OFF_Y2), nullptr, F(OFF_RET), C_DIM, DIN, R_ROWS, F_RES);
    run_stats(F(OFF_Y2), R_ROWS);
    bn_norm<<<R_ROWS, 256, 0, stream>>>(F(OFF_Y2), stats, in[15] + lc, in[16] + lc,
                                        F(OFF_X), BF(OFF_XBF), 1);   // leaky -> nodes

    // ---------- sru ----------
    concat_y18<<<R18_ROWS, 256, 0, stream>>>(BF(OFF_RELAYBF), BF(OFF_XBF), BF(OFF_Y18BF));
    gemm(Wrq, in[18] + lc, BF(OFF_RELAYBF), F(OFF_RQ), nullptr, nullptr, C_DIM, C_DIM, N_BT, 0);
    gemm(Wrk, in[20] + lc, BF(OFF_Y18BF),   F(OFF_RK), nullptr, nullptr, C_DIM, C_DIM, R18_ROWS, 0);
    gemm(Wrv, in[22] + lc, BF(OFF_Y18BF),   F(OFF_RV), nullptr, nullptr, C_DIM, C_DIM, R18_ROWS, 0);
    attn_sru<<<N_BT, 256, 0, stream>>>(F(OFF_RQ), F(OFF_RK), F(OFF_RV),
                                       F(OFF_RELAY), F(OFF_RATT));
    run_stats(F(OFF_RATT), N_BT);
    bn_norm<<<N_BT, 256, 0, stream>>>(F(OFF_RATT), stats, in[23] + lc, in[24] + lc,
                                      F(OFF_RRET), BF(OFF_RRETBF), 0);
    gemm(Wg1, in[26] + ld, BF(OFF_RRETBF), nullptr, BF(OFF_RH), nullptr, DIN, C_DIM, N_BT, F_RELU);
    gemm(Wg2, in[28] + lc, BF(OFF_RH), F(OFF_RY2), nullptr, F(OFF_RRET), C_DIM, DIN, N_BT, F_RES);
    run_stats(F(OFF_RY2), N_BT);
    bn_norm<<<N_BT, 256, 0, stream>>>(F(OFF_RY2), stats, in[29] + lc, in[30] + lc,
                                      F(OFF_RELAY), BF(OFF_RELAYBF), 1);  // leaky -> relay
  }

  output_write<<<out_size / 256, 256, 0, stream>>>(F(OFF_X), (float*)d_out);
}
